// GroupedQueryAttention_2654289789345
// MI455X (gfx1250) — matrix operand order repaired and checked by reading, not compiled, not run on hardware
//
#include <hip/hip_runtime.h>

// B=1, T=2048, DIM=2048, H=16, KV=4, HD=128  (GQA, rep=4)

typedef __attribute__((ext_vector_type(16))) __bf16 v16bf;
typedef __attribute__((ext_vector_type(8)))  float  v8f;
typedef __attribute__((ext_vector_type(4)))  unsigned int v4u;
typedef __attribute__((ext_vector_type(8)))  int v8i;
typedef __attribute__((ext_vector_type(4)))  int v4i;

union AB32 {
  uint4 q[2];
  v16bf v;
};

__device__ __forceinline__ v8f wmma_bf16(v16bf a, v16bf b, v8f c) {
  // (neg_a, A, neg_b, B, c_mod, C, reuse_a, reuse_b)
  return __builtin_amdgcn_wmma_f32_16x16x32_bf16(false, a, false, b, (short)0, c,
                                                 false, false);
}

// ---- A-matrix loader: 16x32 bf16 tile, row-major source with stride ld ----
// ISA layout: lane L holds row m=L&15; lanes 0-15: K=0..7,16..23;
// lanes 16-31: K=8..15,24..31 (per CDNA5 ISA "16-bit A-Matrix 16x32")
__device__ __forceinline__ v16bf load_A_tile(const __bf16* base, int ld) {
  int lane = threadIdx.x & 31;
  int m  = lane & 15;
  int k0 = (lane >> 4) << 3;            // 0 or 8
  const __bf16* p = base + m * ld + k0;
  AB32 u;
  u.q[0] = *reinterpret_cast<const uint4*>(p);       // K = k0..k0+7
  u.q[1] = *reinterpret_cast<const uint4*>(p + 16);  // K = k0+16..k0+23
  return u.v;
}

// ---- B-matrix loader for B = S^T: tile is 32(K) x 16(N), B[k][n]=S[n0+n][k0+k]
// S row-major N x ld. ISA layout: lanes 0-15 hold K=0..7,16..23, lanes 16-31
// K=8..15,24..31; lane's column n = L&15 (same k order as load_A_tile).
__device__ __forceinline__ v16bf load_BT_tile(const __bf16* sbase, int ld) {
  int lane = threadIdx.x & 31;
  int n  = lane & 15;
  int kh = (lane >> 4) << 3;            // 0 or 8
  const __bf16* p = sbase + n * ld + kh;
  AB32 u;
  u.q[0] = *reinterpret_cast<const uint4*>(p);
  u.q[1] = *reinterpret_cast<const uint4*>(p + 16);
  return u.v;
}

// ---- Tensor Data Mover: 2D tile (bf16) global -> LDS ----------------------
// D# packing per CDNA5 ISA (groups 0/1; groups 2/3 and
// the trailing group zeroed for 2D).  dims/strides in 2-byte elements,
// addresses in bytes.  6-arg builtin form (clang-23 toolchain).
__device__ __forceinline__ void tdm_load_2d(unsigned lds_off, const void* gptr,
                                            unsigned dim0, unsigned dim1,
                                            unsigned tile0, unsigned tile1,
                                            unsigned long long stride0) {
  unsigned long long ga = (unsigned long long)(__SIZE_TYPE__)gptr;
  v4u g0;
  g0.x = 1u;                                           // count=1 (valid user D#)
  g0.y = lds_off;                                      // lds_addr (bytes)
  g0.z = (unsigned)(ga & 0xFFFFFFFFu);                 // global_addr[31:0]
  g0.w = (unsigned)((ga >> 32) & 0x01FFFFFFu)          // global_addr[56:32]
         | (2u << 30);                                 // type = 2 ("image")
  v8i g1;
  g1[0] = (int)(1u << 16);                             // data_size=1 -> 2 bytes
  g1[1] = (int)((dim0 & 0xFFFFu) << 16);               // tensor_dim0[15:0]@48
  g1[2] = (int)((dim0 >> 16) | ((dim1 & 0xFFFFu) << 16));
  g1[3] = (int)((dim1 >> 16) | (tile0 << 16));         // tile_dim0@112
  g1[4] = (int)(tile1 & 0xFFFFu);                      // tile_dim1; tile_dim2=0
  g1[5] = (int)(unsigned)(stride0 & 0xFFFFFFFFull);    // tensor_dim0_stride lo
  g1[6] = (int)(unsigned)((stride0 >> 32) & 0xFFFFull);
  g1[7] = 0;                                           // tensor_dim1_stride=0
  v4i z4 = {0, 0, 0, 0};
  v8i z8 = {0, 0, 0, 0, 0, 0, 0, 0};
  __builtin_amdgcn_tensor_load_to_lds(g0, g1, z4, z4, z8, 0);
}

// ======================= elementwise fp32 -> bf16 ==========================
__global__ void cvt_bf16_kernel(const float* __restrict__ src,
                                __bf16* __restrict__ dst, int n) {
  int i = blockIdx.x * blockDim.x + threadIdx.x;
  if (i < n) dst[i] = (__bf16)src[i];
}

// ============ LDS-tiled transpose + convert: src[R,C] f32 -> dst[C,R] bf16 =
__global__ void transpose_cvt_kernel(const float* __restrict__ src,
                                     __bf16* __restrict__ dst, int R, int C) {
  __shared__ float tile[32][33];
  int bc = blockIdx.x * 32;
  int br = blockIdx.y * 32;
  int tx = threadIdx.x & 31;
  int ty = threadIdx.x >> 5;                // 0..7 (blockDim = 256)
#pragma unroll
  for (int i = ty; i < 32; i += 8)
    tile[i][tx] = src[(size_t)(br + i) * C + bc + tx];
  __syncthreads();
#pragma unroll
  for (int i = ty; i < 32; i += 8)
    dst[(size_t)(bc + i) * R + br + tx] = (__bf16)tile[tx][i];
}

// ============ GEMM: Y[M,N] = A[M,Kd] @ W[N,Kd]^T, fp32 out ================
// One wave computes a 16x128 strip; software-pipelined double buffering so
// the loads of slab i+1 overlap the 8 WMMAs of slab i.  launch_bounds(256,1)
// lifts the VGPR cap so both operand buffers stay in registers (no spills).
__global__ void __launch_bounds__(256, 1)
gemm_xwT_kernel(const __bf16* __restrict__ A,
                const __bf16* __restrict__ W,
                float* __restrict__ Y,
                int M, int N, int Kd) {
  int wave    = (blockIdx.x * blockDim.x + threadIdx.x) >> 5;
  int tiles_n = N >> 7;
  int mt = wave / tiles_n;
  int nt = wave % tiles_n;
  if (mt * 16 >= M) return;

  const __bf16* Arow  = A + (size_t)mt * 16 * Kd;
  const __bf16* Wbase = W + (size_t)nt * 128 * Kd;
  v8f acc[8] = {v8f{}, v8f{}, v8f{}, v8f{}, v8f{}, v8f{}, v8f{}, v8f{}};

  v16bf a0 = load_A_tile(Arow, Kd);
  v16bf b0[8];
#pragma unroll
  for (int j = 0; j < 8; ++j) b0[j] = load_BT_tile(Wbase + (size_t)j * 16 * Kd, Kd);

  for (int k = 0; k < Kd; k += 64) {
    v16bf a1 = load_A_tile(Arow + k + 32, Kd);
    v16bf b1[8];
#pragma unroll
    for (int j = 0; j < 8; ++j)
      b1[j] = load_BT_tile(Wbase + (size_t)j * 16 * Kd + k + 32, Kd);
#pragma unroll
    for (int j = 0; j < 8; ++j) acc[j] = wmma_bf16(a0, b0[j], acc[j]);
    if (k + 64 < Kd) {
      a0 = load_A_tile(Arow + k + 64, Kd);
#pragma unroll
      for (int j = 0; j < 8; ++j)
        b0[j] = load_BT_tile(Wbase + (size_t)j * 16 * Kd + k + 64, Kd);
    }
#pragma unroll
    for (int j = 0; j < 8; ++j) acc[j] = wmma_bf16(a1, b1[j], acc[j]);
  }

  int lane = threadIdx.x & 31;
  int n  = lane & 15;
  int hi = lane >> 4;
#pragma unroll
  for (int j = 0; j < 8; ++j) {
#pragma unroll
    for (int r = 0; r < 8; ++r) {
      Y[(size_t)(mt * 16 + r + 8 * hi) * N + nt * 128 + j * 16 + n] = acc[j][r];
    }
  }
}

// ==================== RoPE (+optional scale) fp32 -> bf16 ==================
__global__ void rope_kernel(const float* __restrict__ in,
                            __bf16* __restrict__ out,
                            const float* __restrict__ cosT,
                            const float* __restrict__ sinT,
                            int T, int nheads, float scale) {
  int idx = blockIdx.x * blockDim.x + threadIdx.x;  // T*nheads*64 threads
  int i = idx & 63;
  int h = (idx >> 6) % nheads;
  int t = idx / (64 * nheads);
  if (t >= T) return;
  size_t base = ((size_t)t * nheads + h) * 128;
  float u1 = in[base + i];
  float u2 = in[base + 64 + i];
  float c1 = cosT[t * 128 + i],      s1 = sinT[t * 128 + i];
  float c2 = cosT[t * 128 + 64 + i], s2 = sinT[t * 128 + 64 + i];
  out[base + i]      = (__bf16)((u1 * c1 - u2 * s1) * scale);
  out[base + 64 + i] = (__bf16)((u2 * c2 + u1 * s2) * scale);
}

// ================= flash attention =========================================
// One block = 8 waves = 128 queries of one head (shared KV head g).
// K/V 32-key slabs are staged into LDS by the Tensor Data Mover; waves consume
// them via ds loads.  Qb:[T,16*128] (scale folded), Kb:[T,4*128],
// Vt:[4*128,T] (transposed), Ob:[T,16*128], all bf16.
__global__ void __launch_bounds__(256, 1)
attn_kernel(const __bf16* __restrict__ Qb,
            const __bf16* __restrict__ Kb,
            const __bf16* __restrict__ Vt,
            __bf16* __restrict__ Ob) {
  const int T = 2048, H = 16, KV = 4, HD = 128;
  const int LDQ = H * HD;   // 2048
  const int LDK = KV * HD;  // 512

  __shared__ __align__(16) __bf16 kslab[32 * 128];     // keys x head-dim
  __shared__ __align__(16) __bf16 vslab[128 * 32];     // head-dim x keys
  __shared__ __align__(16) __bf16 lds_p[8][16 * 32];   // per-wave P staging

  int wib  = threadIdx.x >> 5;
  int lane = threadIdx.x & 31;
  int qb0 = blockIdx.x % (T / 128);    // 128-query block
  int h   = blockIdx.x / (T / 128);    // head
  int g   = h / (H / KV);              // kv head (same for whole block)
  int qb  = qb0 * 8 + wib;             // this wave's 16-query block

  const __bf16* Qrow = Qb + (size_t)(qb * 16) * LDQ + h * HD;
  v16bf qa[4];
#pragma unroll
  for (int d = 0; d < 4; ++d) qa[d] = load_A_tile(Qrow + d * 32, LDQ);

  v8f o[8] = {v8f{}, v8f{}, v8f{}, v8f{}, v8f{}, v8f{}, v8f{}, v8f{}};
  float mrow[8], lrow[8];
#pragma unroll
  for (int r = 0; r < 8; ++r) { mrow[r] = -1e30f; lrow[r] = 0.0f; }

  int hi = lane >> 4;
  int nn = lane & 15;
  int my_qmax = qb * 16 + 15;
  int nkb = qb0 * 4 + 4;               // 32-key slabs covering block's qmax
  __bf16* pbuf = &lds_p[wib][0];

  for (int kb = 0; kb < nkb; ++kb) {
    int k0 = kb * 32;
    __syncthreads();                   // previous slab fully consumed
    if (wib == 0) {
      // K slab: 32 rows x 128 cols, row stride LDK elements
      tdm_load_2d((unsigned)(__SIZE_TYPE__)(void*)kslab,
                  Kb + (size_t)k0 * LDK + g * HD, 128, 32, 128, 32,
                  (unsigned long long)LDK);
      // V slab: 128 rows x 32 cols from transposed V, row stride T elements
      tdm_load_2d((unsigned)(__SIZE_TYPE__)(void*)vslab,
                  Vt + (size_t)(g * HD) * T + k0, 32, 128, 32, 128,
                  (unsigned long long)T);
      __builtin_amdgcn_s_wait_tensorcnt(0);
    }
    __syncthreads();                   // slab visible to all waves
    if (k0 > my_qmax) continue;        // past causal limit: barriers only

    // ---- scores: S0 = local keys 0..15, S1 = 16..31 (from LDS) ----
    v8f s0 = v8f{}, s1 = v8f{};
#pragma unroll
    for (int d = 0; d < 4; ++d) {
      v16bf b0 = load_BT_tile(kslab + d * 32, 128);
      v16bf b1 = load_BT_tile(kslab + 16 * 128 + d * 32, 128);
      s0 = wmma_bf16(qa[d], b0, s0);
      s1 = wmma_bf16(qa[d], b1, s1);
    }
    // ---- causal mask (C layout: row m=r+8*hi, col n=nn) ----
#pragma unroll
    for (int r = 0; r < 8; ++r) {
      int qm = qb * 16 + r + 8 * hi;
      if (k0 + nn > qm)      s0[r] = -1e30f;
      if (k0 + 16 + nn > qm) s1[r] = -1e30f;
    }
    // ---- row max across the 16 lanes of this half ----
    float tm[8];
#pragma unroll
    for (int r = 0; r < 8; ++r) tm[r] = fmaxf(s0[r], s1[r]);
#pragma unroll
    for (int off = 1; off < 16; off <<= 1)
#pragma unroll
      for (int r = 0; r < 8; ++r) tm[r] = fmaxf(tm[r], __shfl_xor(tm[r], off, 32));

    float alpha[8];
#pragma unroll
    for (int r = 0; r < 8; ++r) {
      float mnew = fmaxf(mrow[r], tm[r]);
      alpha[r] = __expf(mrow[r] - mnew);
      mrow[r] = mnew;
    }
    float rs[8];
#pragma unroll
    for (int r = 0; r < 8; ++r) {
      s0[r] = __expf(s0[r] - mrow[r]);
      s1[r] = __expf(s1[r] - mrow[r]);
      rs[r] = s0[r] + s1[r];
    }
#pragma unroll
    for (int off = 1; off < 16; off <<= 1)
#pragma unroll
      for (int r = 0; r < 8; ++r) rs[r] += __shfl_xor(rs[r], off, 32);
#pragma unroll
    for (int r = 0; r < 8; ++r) lrow[r] = lrow[r] * alpha[r] + rs[r];
#pragma unroll
    for (int t = 0; t < 8; ++t)
#pragma unroll
      for (int r = 0; r < 8; ++r) o[t][r] *= alpha[r];

    // ---- stage P (16x32) to LDS, reload in A layout ----
#pragma unroll
    for (int r = 0; r < 8; ++r) {
      pbuf[(r + 8 * hi) * 32 + nn]      = (__bf16)s0[r];
      pbuf[(r + 8 * hi) * 32 + 16 + nn] = (__bf16)s1[r];
    }
    asm volatile("s_wait_dscnt 0x0" ::: "memory");
    v16bf pa = load_A_tile(pbuf, 32);
    asm volatile("s_wait_dscnt 0x0" ::: "memory");

    // ---- O += P(16x32) @ V(32x128), V slab dim-major in LDS (ld=32) ----
#pragma unroll
    for (int t = 0; t < 8; ++t) {
      v16bf vb = load_BT_tile(vslab + t * 16 * 32, 32);
      o[t] = wmma_bf16(pa, vb, o[t]);
    }
  }

  // ---- epilogue: normalize, store bf16 ----
#pragma unroll
  for (int t = 0; t < 8; ++t) {
#pragma unroll
    for (int r = 0; r < 8; ++r) {
      float val = o[t][r] / lrow[r];
      Ob[(size_t)(qb * 16 + r + 8 * hi) * LDQ + h * HD + t * 16 + nn] = (__bf16)val;
    }
  }
}

// ============================ host launcher ================================
extern "C" void kernel_launch(void* const* d_in, const int* in_sizes, int n_in,
                              void* d_out, int out_size, void* d_ws, size_t ws_size,
                              hipStream_t stream) {
  (void)in_sizes; (void)n_in; (void)out_size; (void)ws_size;
  const int T = 2048, DIM = 2048, H = 16, KV = 4, HD = 128;
  const float scale = 0.08838834764831845f;  // 128^-0.5

  const float* x    = (const float*)d_in[0];
  const float* wq   = (const float*)d_in[1];
  const float* wk   = (const float*)d_in[2];
  const float* wv   = (const float*)d_in[3];
  const float* wo   = (const float*)d_in[4];
  const float* cosT = (const float*)d_in[5];
  const float* sinT = (const float*)d_in[6];
  float* out = (float*)d_out;

  // workspace carve (~76 MB total)
  char* w = (char*)d_ws;
  __bf16* xb  = (__bf16*)w; w += (size_t)T * DIM * 2;
  __bf16* wqb = (__bf16*)w; w += (size_t)H * HD * DIM * 2;
  __bf16* wkb = (__bf16*)w; w += (size_t)KV * HD * DIM * 2;
  __bf16* wvb = (__bf16*)w; w += (size_t)KV * HD * DIM * 2;
  __bf16* wob = (__bf16*)w; w += (size_t)DIM * H * HD * 2;
  float*  Qf  = (float*)w;  w += (size_t)T * H * HD * 4;
  float*  Kf  = (float*)w;  w += (size_t)T * KV * HD * 4;
  float*  Vf  = (float*)w;  w += (size_t)T * KV * HD * 4;
  __bf16* Qr  = (__bf16*)w; w += (size_t)T * H * HD * 2;
  __bf16* Kr  = (__bf16*)w; w += (size_t)T * KV * HD * 2;
  __bf16* Vt  = (__bf16*)w; w += (size_t)KV * HD * T * 2;   // transposed V
  __bf16* Ao  = (__bf16*)w; w += (size_t)T * H * HD * 2;

  auto cvt = [&](const float* s, __bf16* d, int n) {
    hipLaunchKernelGGL(cvt_bf16_kernel, dim3((n + 255) / 256), dim3(256), 0, stream,
                       s, d, n);
  };
  cvt(x,  xb,  T * DIM);
  cvt(wq, wqb, H * HD * DIM);
  cvt(wk, wkb, KV * HD * DIM);
  cvt(wv, wvb, KV * HD * DIM);
  cvt(wo, wob, DIM * H * HD);

  auto gemm = [&](const __bf16* A, const __bf16* W_, float* Y, int M, int N, int Kd) {
    int waves = (M / 16) * (N / 128);
    hipLaunchKernelGGL(gemm_xwT_kernel, dim3(waves / 8), dim3(256), 0, stream,
                       A, W_, Y, M, N, Kd);
  };
  gemm(xb, wqb, Qf, T, H * HD, DIM);    // Q = x @ wq^T
  gemm(xb, wkb, Kf, T, KV * HD, DIM);   // K = x @ wk^T
  gemm(xb, wvb, Vf, T, KV * HD, DIM);   // V = x @ wv^T

  hipLaunchKernelGGL(rope_kernel, dim3(T * H * 64 / 256), dim3(256), 0, stream,
                     Qf, Qr, cosT, sinT, T, H, scale);
  hipLaunchKernelGGL(rope_kernel, dim3(T * KV * 64 / 256), dim3(256), 0, stream,
                     Kf, Kr, cosT, sinT, T, KV, 1.0f);

  // V: fp32 [T, KV*HD] -> bf16 transposed [KV*HD, T]
  hipLaunchKernelGGL(transpose_cvt_kernel, dim3(KV * HD / 32, T / 32), dim3(256),
                     0, stream, Vf, Vt, T, KV * HD);

  int ablocks = H * (T / 128);  // 256 blocks x 8 waves
  hipLaunchKernelGGL(attn_kernel, dim3(ablocks), dim3(256), 0, stream,
                     Qr, Kr, Vt, Ao);

  gemm(Ao, wob, out, T, DIM, DIM);      // out = attn @ wo^T
}
